// FeatureGenerator_6279242186811
// MI455X (gfx1250) — matrix operand order repaired and checked by reading, not compiled, not run on hardware
//
#include <hip/hip_runtime.h>

// ---------------------------------------------------------------------------
// Problem constants (from the reference)
// ---------------------------------------------------------------------------
#define BATCH   4
#define NPTS    120000
#define NT      480000          // BATCH*NPTS, also V (output rows)
#define GRID    128
#define DENSE   8388608         // BATCH*GRID^3  (= 2^23)
#define EPSBN   1e-5f

typedef __attribute__((ext_vector_type(16))) _Float16 v16h;
typedef __attribute__((ext_vector_type(8)))  _Float16 v8h;
typedef __attribute__((ext_vector_type(4)))  _Float16 v4h;
typedef __attribute__((ext_vector_type(8)))  float    v8f;

// ---------------------------------------------------------------------------
// Utility: fill u32 pattern (zeroing accumulators / -inf seg init)
// ---------------------------------------------------------------------------
__global__ void fill_u32_kernel(unsigned* __restrict__ p, long long n, unsigned val) {
  long long i = (long long)blockIdx.x * blockDim.x + threadIdx.x;
  long long stride = (long long)gridDim.x * blockDim.x;
  for (; i < n; i += stride) p[i] = val;
}

// ---------------------------------------------------------------------------
// Voxel key marking: key = ((b*G + x)*G + y)*G + z ; presence[key] = 1
// ---------------------------------------------------------------------------
__global__ void mark_kernel(const int* __restrict__ coord,
                            int* __restrict__ keys,
                            unsigned* __restrict__ pres) {
  int p = blockIdx.x * blockDim.x + threadIdx.x;
  if (p >= NT) return;
  int x = coord[3 * p + 0], y = coord[3 * p + 1], z = coord[3 * p + 2];
  int b = p / NPTS;
  int key = ((b * GRID + x) * GRID + y) * GRID + z;
  keys[p] = key;
  pres[key] = 1u;
}

// ---------------------------------------------------------------------------
// Hierarchical prefix scan over 8.4M presence flags (2048 blocks x 4096 elems)
// rank[key] = #present keys < key == index in sorted-unique order (jnp.unique)
// ---------------------------------------------------------------------------
__global__ void scan1_kernel(const unsigned* __restrict__ pres,
                             unsigned* __restrict__ bsum) {
  __shared__ unsigned ls[1024];
  int t = threadIdx.x;
  size_t base = (size_t)blockIdx.x * 4096 + (size_t)t * 4;
  uint4 v = *(const uint4*)(pres + base);
  ls[t] = v.x + v.y + v.z + v.w;
  __syncthreads();
  for (int off = 512; off > 0; off >>= 1) {
    if (t < off) ls[t] += ls[t + off];
    __syncthreads();
  }
  if (t == 0) bsum[blockIdx.x] = ls[0];
}

__global__ void scan2_kernel(const unsigned* __restrict__ bsum,
                             unsigned* __restrict__ boff,
                             unsigned* __restrict__ Ucnt) {
  __shared__ unsigned ls[1024];
  int t = threadIdx.x;
  unsigned s0 = bsum[2 * t], s1 = bsum[2 * t + 1];
  unsigned pr = s0 + s1;
  ls[t] = pr;
  __syncthreads();
  for (int off = 1; off < 1024; off <<= 1) {
    unsigned x = (t >= off) ? ls[t - off] : 0u;
    __syncthreads();
    ls[t] += x;
    __syncthreads();
  }
  unsigned excl = ls[t] - pr;
  boff[2 * t] = excl;
  boff[2 * t + 1] = excl + s0;
  if (t == 1023) *Ucnt = ls[1023];
}

__global__ void scan3_kernel(const unsigned* __restrict__ pres,
                             const unsigned* __restrict__ boff,
                             unsigned* __restrict__ rank) {
  __shared__ unsigned ls[1024];
  int t = threadIdx.x;
  size_t base = (size_t)blockIdx.x * 4096 + (size_t)t * 4;
  uint4 v = *(const uint4*)(pres + base);
  unsigned s = v.x + v.y + v.z + v.w;
  ls[t] = s;
  __syncthreads();
  for (int off = 1; off < 1024; off <<= 1) {
    unsigned x = (t >= off) ? ls[t - off] : 0u;
    __syncthreads();
    ls[t] += x;
    __syncthreads();
  }
  unsigned thr = boff[blockIdx.x] + ls[t] - s;   // exclusive prefix at thread start
  uint4 o;
  o.x = thr;
  o.y = thr + v.x;
  o.z = o.y + v.y;
  o.w = o.z + v.z;
  *(uint4*)(rank + base) = o;
}

__global__ void inv_kernel(const int* __restrict__ keys,
                           const unsigned* __restrict__ rank,
                           int* __restrict__ invp) {
  int p = blockIdx.x * blockDim.x + threadIdx.x;
  if (p >= NT) return;
  invp[p] = (int)rank[keys[p]];
}

// unique_coords: present voxels -> sorted slot; pads -> (-1,127,127,127)
__global__ void coords_present_kernel(const unsigned* __restrict__ pres,
                                      const unsigned* __restrict__ rank,
                                      int* __restrict__ co) {
  int k = blockIdx.x * blockDim.x + threadIdx.x;
  if (k >= DENSE) return;
  if (!pres[k]) return;
  unsigned r = rank[k];
  int* o = co + ((size_t)r << 2);
  o[0] = k >> 21;
  o[1] = (k >> 14) & 127;
  o[2] = (k >> 7) & 127;
  o[3] = k & 127;
}

__global__ void coords_pad_kernel(int* __restrict__ co,
                                  const unsigned* __restrict__ Ucnt) {
  int r = blockIdx.x * blockDim.x + threadIdx.x;
  if (r >= NT) return;
  if ((unsigned)r < *Ucnt) return;
  int* o = co + ((size_t)r << 2);
  o[0] = -1; o[1] = 127; o[2] = 127; o[3] = 127;
}

// ---------------------------------------------------------------------------
// Per-column batch statistics.
// Generic atomic version (only used for the tiny 9-column bn0 pass).
// ---------------------------------------------------------------------------
template <typename T>
__global__ void stats_atomic_kernel(const T* __restrict__ y, long long total, int N,
                                    float* __restrict__ sum, float* __restrict__ sq) {
  __shared__ float ls[512];
  for (int i = threadIdx.x; i < 2 * N; i += blockDim.x) ls[i] = 0.0f;
  __syncthreads();
  long long idx = (long long)blockIdx.x * blockDim.x + threadIdx.x;
  long long stride = (long long)gridDim.x * blockDim.x;
  for (; idx < total; idx += stride) {
    float v = (float)y[idx];
    int c = (int)(idx % N);
    atomicAdd(&ls[c], v);
    atomicAdd(&ls[N + c], v * v);
  }
  __syncthreads();
  for (int c = threadIdx.x; c < N; c += blockDim.x) {
    atomicAdd(&sum[c], ls[c]);
    atomicAdd(&sq[c], ls[N + c]);
  }
}

// Register-accumulating version for N in {64,128,256}: thread owns one column,
// coalesced reads, LDS tree-reduce across row-subgroups, N atomics per block.
template <int N>
__global__ void stats_reg_kernel(const _Float16* __restrict__ y, int M,
                                 float* __restrict__ sum, float* __restrict__ sq) {
  __shared__ float ls[512];
  const int tid = threadIdx.x;
  const int c = tid % N;
  const int rsub = tid / N;              // 0 .. 256/N - 1
  const int rowsPerBlock = M / gridDim.x;
  const int r0 = blockIdx.x * rowsPerBlock;
  float s = 0.0f, s2 = 0.0f;
  for (int r = r0 + rsub; r < r0 + rowsPerBlock; r += 256 / N) {
    float v = (float)y[(size_t)r * N + c];
    s += v;
    s2 += v * v;
  }
  ls[tid] = s;
  ls[256 + tid] = s2;
  __syncthreads();
  for (int off = 128; off >= N; off >>= 1) {
    if (tid < off) {
      ls[tid] += ls[tid + off];
      ls[256 + tid] += ls[256 + tid + off];
    }
    __syncthreads();
  }
  if (tid < N) {
    atomicAdd(&sum[tid], ls[tid]);
    atomicAdd(&sq[tid], ls[256 + tid]);
  }
}

__global__ void finalize_stats_kernel(const float* __restrict__ sum,
                                      const float* __restrict__ sq,
                                      const float* __restrict__ gamma,
                                      const float* __restrict__ beta,
                                      float* __restrict__ scale,
                                      float* __restrict__ shift,
                                      int N, float invM) {
  int c = blockIdx.x * blockDim.x + threadIdx.x;
  if (c >= N) return;
  float m = sum[c] * invM;
  float var = sq[c] * invM - m * m;
  float s = gamma[c] * rsqrtf(var + EPSBN);
  scale[c] = s;
  shift[c] = beta[c] - m * s;
}

// a = relu(y*scale + shift), v8h (128-bit) vectorized; scale/shift LDS-cached
__global__ void normalize_kernel(const _Float16* __restrict__ y,
                                 _Float16* __restrict__ a,
                                 const float* __restrict__ scale,
                                 const float* __restrict__ shift,
                                 long long total8, int N) {
  __shared__ float lsc[256], lsh[256];
  for (int i = threadIdx.x; i < N; i += blockDim.x) {
    lsc[i] = scale[i];
    lsh[i] = shift[i];
  }
  __syncthreads();
  long long idx = (long long)blockIdx.x * blockDim.x + threadIdx.x;
  long long stride = (long long)gridDim.x * blockDim.x;
  for (; idx < total8; idx += stride) {
    v8h v = *(const v8h*)(y + idx * 8);
    int c0 = (int)((idx * 8) % N);       // N % 8 == 0 -> no wrap within vector
    v8h o;
#pragma unroll
    for (int j = 0; j < 8; ++j) {
      float f = (float)v[j] * lsc[c0 + j] + lsh[c0 + j];
      o[j] = (_Float16)fmaxf(f, 0.0f);
    }
    *(v8h*)(a + idx * 8) = o;
  }
}

// ---------------------------------------------------------------------------
// Layer1 (K=9, tiny): y1 = bn0(feat) @ W1 + b1, weights cached in LDS
// ---------------------------------------------------------------------------
__global__ void layer1_kernel(const float* __restrict__ feat,
                              const float* __restrict__ scale0,
                              const float* __restrict__ shift0,
                              const float* __restrict__ W1,
                              const float* __restrict__ b1,
                              _Float16* __restrict__ y) {
  __shared__ float w[9 * 64];
  __shared__ float bb[64];
  __shared__ float sc[9], sh[9];
  for (int i = threadIdx.x; i < 576; i += 256) w[i] = W1[i];
  if (threadIdx.x < 64) bb[threadIdx.x] = b1[threadIdx.x];
  if (threadIdx.x < 9) { sc[threadIdx.x] = scale0[threadIdx.x]; sh[threadIdx.x] = shift0[threadIdx.x]; }
  __syncthreads();
  int p = blockIdx.x * 256 + threadIdx.x;   // NT/256 = 1875 exact
  float xn[9];
#pragma unroll
  for (int i = 0; i < 9; ++i) xn[i] = feat[(size_t)p * 9 + i] * sc[i] + sh[i];
  _Float16* yr = y + (size_t)p * 64;
#pragma unroll 4
  for (int j = 0; j < 64; ++j) {
    float acc = bb[j];
#pragma unroll
    for (int i = 0; i < 9; ++i) acc += xn[i] * w[i * 64 + j];
    yr[j] = (_Float16)acc;
  }
}

// ---------------------------------------------------------------------------
// Weight repack: row-major f32 W[K,N] -> fragment-major f16, kc-major:
//   Wp[ ((kc*(N/16) + nt)*32 + lane)*16 + e ]
// B element e at lane (g = lane>>4, n = lane&15): k = kc*32 + 8*g + e (e<8), kc*32 + 16 + 8*g + (e-8) (e>=8), col = nt*16+n
// kc-major makes the NTILES fragments of one k-step contiguous -> constant
// immediate offsets in the GEMM inner loop.
// ---------------------------------------------------------------------------
__global__ void repack_w_kernel(const float* __restrict__ W,
                                _Float16* __restrict__ Wp, int K, int N) {
  int idx = blockIdx.x * blockDim.x + threadIdx.x;
  int kcn = K >> 5;
  int ntt = N >> 4;
  int total = ntt * kcn * 32;
  if (idx >= total) return;
  int lane = idx & 31;
  int rest = idx >> 5;
  int nt = rest % ntt;
  int kc = rest / ntt;
  int g = lane >> 4, n = lane & 15;
  _Float16* dst = Wp + ((size_t)idx << 4);
  int kbase = (kc << 5) + (g << 3);
  int col = (nt << 4) + n;
#pragma unroll
  for (int e = 0; e < 16; ++e)
    dst[e] = (_Float16)W[(size_t)(kbase + e + ((e >> 3) << 3)) * N + col];
}

// ---------------------------------------------------------------------------
// WMMA GEMM: Y[M,N] = A[M,K](f16) @ Wp(f16, kc-major fragments) + bias
// One wave computes a 16 x (16*NTILES) strip. N and K are compile-time so the
// K loop fully unrolls and all loads are base + constant immediate offset;
// B fragments are loaded into distinct registers before the WMMA batch so the
// scheduler can keep loads in flight (partial s_wait_loadcnt).
// 8 waves/block cover 128 rows; K multiple of 32.
// ---------------------------------------------------------------------------
template <int NTILES, int N, int K, bool RELU_OUT, bool OUT_F32>
__global__ void wmma_gemm_kernel(const _Float16* __restrict__ A,
                                 const _Float16* __restrict__ Bp,
                                 const float* __restrict__ bias,
                                 _Float16* __restrict__ Yh,
                                 float* __restrict__ Yf,
                                 int M) {
  constexpr int KCN = K >> 5;
  constexpr int NTT = N >> 4;
  const int lane = threadIdx.x & 31;
  const int wave = threadIdx.x >> 5;
  const int m0 = (blockIdx.x * 8 + wave) << 4;
  const int nt0 = blockIdx.y * NTILES;
  const int g = lane >> 4;
  const int mr = lane & 15;
  v8f acc[NTILES];
#pragma unroll
  for (int t = 0; t < NTILES; ++t) acc[t] = (v8f){};
  // A fragment: lane (g,mr) element e: e<8 -> k=k0+8g+e ; e>=8 -> k=k0+16+8g+(e-8)
  const _Float16* arow = A + (size_t)(m0 + mr) * K + (g << 3);
  const _Float16* bbase = Bp + (((size_t)nt0 * 32 + lane) << 4);
#pragma unroll
  for (int kc = 0; kc < KCN; ++kc) {
    v8h alo = *(const v8h*)(arow + kc * 32);
    v8h ahi = *(const v8h*)(arow + kc * 32 + 16);
    v16h a;
#pragma unroll
    for (int i = 0; i < 8; ++i) { a[i] = alo[i]; a[i + 8] = ahi[i]; }
    v16h b[NTILES];
#pragma unroll
    for (int t = 0; t < NTILES; ++t)
      b[t] = *(const v16h*)(bbase + ((size_t)kc * NTT + t) * 512);
#pragma unroll
    for (int t = 0; t < NTILES; ++t)
      acc[t] = __builtin_amdgcn_wmma_f32_16x16x32_f16(false, a, false, b[t],
                                                      (short)0, acc[t], false, false);
  }
#pragma unroll
  for (int t = 0; t < NTILES; ++t) {
    const int n0 = (nt0 + t) << 4;
    const float bv = bias[n0 + mr];
#pragma unroll
    for (int r = 0; r < 8; ++r) {
      int row = m0 + (g << 3) + r;       // C/D: VGPR r -> M = r + 8*g
      float v = acc[t][r] + bv;
      if (RELU_OUT) v = fmaxf(v, 0.0f);
      size_t o = (size_t)row * N + n0 + mr;
      if (OUT_F32) Yf[o] = v; else Yh[o] = (_Float16)v;
    }
  }
}

// ---------------------------------------------------------------------------
// segment_max via sign-split f32 atomic max (seg pre-initialized to -inf)
// ---------------------------------------------------------------------------
__device__ __forceinline__ void atomicMaxF32(float* addr, float v) {
  if (v >= 0.0f) atomicMax((int*)addr, __float_as_int(v));
  else           atomicMin((unsigned int*)addr, __float_as_uint(v));
}

__global__ void scatter_max_kernel(const _Float16* __restrict__ y4,
                                   const int* __restrict__ invp,
                                   float* __restrict__ seg) {
  long long idx = (long long)blockIdx.x * 256 + threadIdx.x;  // grid = NT blocks
  int p = (int)(idx >> 8);
  int c = (int)(idx & 255);
  int r = invp[p];
  atomicMaxF32(&seg[((size_t)r << 8) + c], (float)y4[idx]);
}

// classifier input: f16(seg) for rows < U, zero for padded rows (=> relu(bc))
__global__ void seg_to_f16_kernel(const float* __restrict__ seg,
                                  _Float16* __restrict__ a,
                                  const unsigned* __restrict__ Ucnt) {
  long long i4 = (long long)blockIdx.x * 256 + threadIdx.x;   // units of 4 floats
  unsigned U = *Ucnt;
  unsigned r = (unsigned)((i4 * 4) >> 8);
  float4 v = *(const float4*)(seg + i4 * 4);
  v4h o;
  if (r < U) {
    o[0] = (_Float16)v.x; o[1] = (_Float16)v.y;
    o[2] = (_Float16)v.z; o[3] = (_Float16)v.w;
  } else {
    o[0] = (_Float16)0.0f; o[1] = (_Float16)0.0f;
    o[2] = (_Float16)0.0f; o[3] = (_Float16)0.0f;
  }
  *(v4h*)(a + i4 * 4) = o;
}

// ---------------------------------------------------------------------------
// Host launcher
// ---------------------------------------------------------------------------
extern "C" void kernel_launch(void* const* d_in, const int* in_sizes, int n_in,
                              void* d_out, int out_size, void* d_ws, size_t ws_size,
                              hipStream_t stream) {
  (void)in_sizes; (void)n_in; (void)out_size; (void)ws_size;

  const float* feat  = (const float*)d_in[0];
  const int*   coord = (const int*)  d_in[1];
  const float* bn0_g = (const float*)d_in[2];
  const float* bn0_b = (const float*)d_in[3];
  const float* w1    = (const float*)d_in[4];
  const float* b1    = (const float*)d_in[5];
  const float* bn1_g = (const float*)d_in[6];
  const float* bn1_b = (const float*)d_in[7];
  const float* w2    = (const float*)d_in[8];
  const float* b2    = (const float*)d_in[9];
  const float* bn2_g = (const float*)d_in[10];
  const float* bn2_b = (const float*)d_in[11];
  const float* w3    = (const float*)d_in[12];
  const float* b3    = (const float*)d_in[13];
  const float* bn3_g = (const float*)d_in[14];
  const float* bn3_b = (const float*)d_in[15];
  const float* w4    = (const float*)d_in[16];
  const float* b4    = (const float*)d_in[17];
  const float* wc    = (const float*)d_in[18];
  const float* bc    = (const float*)d_in[19];

  float* out = (float*)d_out;                           // [NT,16] f32
  int*   co  = (int*)((float*)d_out + (size_t)NT * 16); // [NT,4] int32 (raw bits)

  // bump allocator over workspace
  char* ws = (char*)d_ws;
  size_t off = 0;
  auto alloc = [&](size_t bytes) -> void* {
    void* p = ws + off;
    off = (off + bytes + 255) & ~(size_t)255;
    return p;
  };
  _Float16* bufY = (_Float16*)alloc((size_t)NT * 256 * 2); // raw layer outputs (f16)
  _Float16* bufA = (_Float16*)alloc((size_t)NT * 256 * 2); // normalized activations
  float*    seg  = (float*)   alloc((size_t)NT * 256 * 4); // segment-max accum
  unsigned* pres = (unsigned*)alloc((size_t)DENSE * 4);
  unsigned* rank = (unsigned*)alloc((size_t)DENSE * 4);
  int*      keys = (int*)     alloc((size_t)NT * 4);
  int*      invp = (int*)     alloc((size_t)NT * 4);
  unsigned* bsum = (unsigned*)alloc(2048 * 4);
  unsigned* boff = (unsigned*)alloc(2048 * 4);
  unsigned* Ucnt = (unsigned*)alloc(256);
  float*    ssum = (float*)   alloc(256 * 4);
  float*    ssq  = (float*)   alloc(256 * 4);
  float*    scl  = (float*)   alloc(256 * 4);
  float*    shf  = (float*)   alloc(256 * 4);
  _Float16* Wp   = (_Float16*)alloc((size_t)256 * 256 * 2);

  // ---- Phase 1: voxel unique / inverse / coords (independent of MLP) ----
  fill_u32_kernel<<<4096, 256, 0, stream>>>(pres, (long long)DENSE, 0u);
  mark_kernel<<<NT / 256, 256, 0, stream>>>(coord, keys, pres);
  scan1_kernel<<<2048, 1024, 0, stream>>>(pres, bsum);
  scan2_kernel<<<1, 1024, 0, stream>>>(bsum, boff, Ucnt);
  scan3_kernel<<<2048, 1024, 0, stream>>>(pres, boff, rank);
  inv_kernel<<<NT / 256, 256, 0, stream>>>(keys, rank, invp);
  coords_present_kernel<<<DENSE / 256, 256, 0, stream>>>(pres, rank, co);
  coords_pad_kernel<<<NT / 256, 256, 0, stream>>>(co, Ucnt);

  const float invM = 1.0f / (float)NT;
  auto zero_stats = [&]() {
    fill_u32_kernel<<<1, 256, 0, stream>>>((unsigned*)ssum, 256, 0u);
    fill_u32_kernel<<<1, 256, 0, stream>>>((unsigned*)ssq, 256, 0u);
  };

  // ---- Phase 2: MLP ----
  // bn0 on raw features, folded into layer1
  zero_stats();
  stats_atomic_kernel<float><<<1024, 256, 0, stream>>>(feat, (long long)NT * 9, 9,
                                                       ssum, ssq);
  finalize_stats_kernel<<<1, 256, 0, stream>>>(ssum, ssq, bn0_g, bn0_b, scl, shf, 9, invM);
  layer1_kernel<<<NT / 256, 256, 0, stream>>>(feat, scl, shf, w1, b1, bufY);

  // bn1 + relu -> a1 ; layer2 (64 -> 128)
  zero_stats();
  stats_reg_kernel<64><<<480, 256, 0, stream>>>(bufY, NT, ssum, ssq);
  finalize_stats_kernel<<<1, 256, 0, stream>>>(ssum, ssq, bn1_g, bn1_b, scl, shf, 64, invM);
  normalize_kernel<<<8192, 256, 0, stream>>>(bufY, bufA, scl, shf,
                                             (long long)NT * 64 / 8, 64);
  repack_w_kernel<<<2, 256, 0, stream>>>(w2, Wp, 64, 128);          // (128/16)*(64/32)*32 = 512
  wmma_gemm_kernel<4, 128, 64, false, false><<<dim3(NT / 128, 2), 256, 0, stream>>>(
      bufA, Wp, b2, bufY, nullptr, NT);

  // bn2 + relu -> a2 ; layer3 (128 -> 256)
  zero_stats();
  stats_reg_kernel<128><<<480, 256, 0, stream>>>(bufY, NT, ssum, ssq);
  finalize_stats_kernel<<<1, 256, 0, stream>>>(ssum, ssq, bn2_g, bn2_b, scl, shf, 128, invM);
  normalize_kernel<<<8192, 256, 0, stream>>>(bufY, bufA, scl, shf,
                                             (long long)NT * 128 / 8, 128);
  repack_w_kernel<<<8, 256, 0, stream>>>(w3, Wp, 128, 256);         // 16*4*32 = 2048
  wmma_gemm_kernel<4, 256, 128, false, false><<<dim3(NT / 128, 4), 256, 0, stream>>>(
      bufA, Wp, b3, bufY, nullptr, NT);

  // bn3 + relu -> a3 ; layer4 (256 -> 256, no bn/relu after)
  zero_stats();
  stats_reg_kernel<256><<<480, 256, 0, stream>>>(bufY, NT, ssum, ssq);
  finalize_stats_kernel<<<1, 256, 0, stream>>>(ssum, ssq, bn3_g, bn3_b, scl, shf, 256, invM);
  normalize_kernel<<<8192, 256, 0, stream>>>(bufY, bufA, scl, shf,
                                             (long long)NT * 256 / 8, 256);
  repack_w_kernel<<<16, 256, 0, stream>>>(w4, Wp, 256, 256);        // 16*8*32 = 4096
  wmma_gemm_kernel<4, 256, 256, false, false><<<dim3(NT / 128, 4), 256, 0, stream>>>(
      bufA, Wp, b4, bufY, nullptr, NT);                             // y4 (f16)

  // ---- Phase 3: segment max + classifier ----
  fill_u32_kernel<<<8192, 256, 0, stream>>>((unsigned*)seg, (long long)NT * 256,
                                            0xFF800000u);           // -inf
  scatter_max_kernel<<<NT, 256, 0, stream>>>(bufY, invp, seg);
  seg_to_f16_kernel<<<NT / 4, 256, 0, stream>>>(seg, bufA, Ucnt);
  repack_w_kernel<<<1, 256, 0, stream>>>(wc, Wp, 256, 16);          // 1*8*32 = 256
  wmma_gemm_kernel<1, 16, 256, true, true><<<dim3(NT / 128, 1), 256, 0, stream>>>(
      bufA, Wp, bc, nullptr, out, NT);
}
